// AllAtomEnergyBranch_32169305047137
// MI455X (gfx1250) — matrix operand order repaired and checked by reading, not compiled, not run on hardware
//
#include <hip/hip_runtime.h>

typedef __attribute__((ext_vector_type(16))) __bf16 v16bf;
typedef __attribute__((ext_vector_type(8)))  float  v8f;

#define HDIM 256   // H
#define EDIM 64    // E
#define HID  512   // hidden
#define NE   256   // nE
#define BN   512   // B*N

// ---------------- prep 1: hp[bn][k] = h_all[bn,:] @ Wh[:,k]  (fp32) --------
__global__ __launch_bounds__(256) void prep_hpart(const float* __restrict__ h_all,
                                                  const float* __restrict__ W1,
                                                  float* __restrict__ hp) {
  __shared__ float hsh[8][HDIM];
  int bn0 = blockIdx.x * 8;
  for (int i = threadIdx.x; i < 8 * HDIM; i += 256) {
    int r = i / HDIM, c = i % HDIM;
    hsh[r][c] = h_all[(bn0 + r) * HDIM + c];
  }
  __syncthreads();
  int bl = threadIdx.x >> 5;          // local bn row (0..7)
  int k0 = (threadIdx.x & 31) * 16;   // 16 output cols per thread
  float acc[16];
#pragma unroll
  for (int i = 0; i < 16; ++i) acc[i] = 0.f;
  for (int r = 0; r < HDIM; ++r) {
    float h = hsh[bl][r];
    const float* w = W1 + r * HID + k0;   // Wh rows are W1[0..255]
#pragma unroll
    for (int i = 0; i < 16; ++i) acc[i] += h * w[i];
  }
  float* o = hp + (bn0 + bl) * HID + k0;
#pragma unroll
  for (int i = 0; i < 16; ++i) o[i] = acc[i];
}

// ---------------- prep 2: ep[e][k] = e_feat[e,:] @ We[:,k] + b1[k] ---------
__global__ __launch_bounds__(256) void prep_epart(const float* __restrict__ e_feat,
                                                  const float* __restrict__ W1,
                                                  const float* __restrict__ b1,
                                                  float* __restrict__ ep) {
  __shared__ float esh[8][EDIM];
  int e0 = blockIdx.x * 8;
  for (int i = threadIdx.x; i < 8 * EDIM; i += 256) {
    int r = i / EDIM, c = i % EDIM;
    esh[r][c] = e_feat[(e0 + r) * EDIM + c];
  }
  __syncthreads();
  int el = threadIdx.x >> 5;
  int k0 = (threadIdx.x & 31) * 16;
  float acc[16];
#pragma unroll
  for (int i = 0; i < 16; ++i) acc[i] = b1[k0 + i];
  const float* We = W1 + HDIM * HID;      // We rows are W1[256..319]
  for (int r = 0; r < EDIM; ++r) {
    float v = esh[el][r];
    const float* w = We + r * HID + k0;
#pragma unroll
    for (int i = 0; i < 16; ++i) acc[i] += v * w[i];
  }
  float* o = ep + (e0 + el) * HID + k0;
#pragma unroll
  for (int i = 0; i < 16; ++i) o[i] = acc[i];
}

// ---------------- prep 3: W2T[n][k] = bf16(W2[k][n]) -----------------------
__global__ __launch_bounds__(256) void prep_w2t(const float* __restrict__ W2,
                                                __bf16* __restrict__ W2T) {
  int idx = blockIdx.x * 256 + threadIdx.x;   // 512*512 elements
  int k = idx >> 9, n = idx & 511;
  W2T[n * HID + k] = (__bf16)W2[k * HID + n];
}

// --------- CDNA5 async global -> LDS (16B per lane), tracked by ASYNCcnt ---
__device__ __forceinline__ void async_b128(unsigned lds_byte, unsigned goff,
                                           const void* sbase) {
  asm volatile("global_load_async_to_lds_b128 %0, %1, %2"
               :: "v"(lds_byte), "v"(goff), "s"(sbase)
               : "memory");
}

// ---------------- main fused kernel ----------------------------------------
// wave handles a 16-row A-tile (16 e values for one bn) across all 512 cols.
// B tiles (shared by all 8 waves of the block) are double-buffered in LDS,
// filled via the async-to-LDS engine in a fragment-swizzled layout.
__global__ __launch_bounds__(256, 1) void fused_mlp(
    const float* __restrict__ hp, const float* __restrict__ ep,
    const __bf16* __restrict__ W2T, const float* __restrict__ b2,
    const float* __restrict__ W3, const float* __restrict__ b3,
    float* __restrict__ out) {
  __shared__ __bf16 Bsh[2][8192];           // 2 x 16KB B tiles

  int lane = threadIdx.x & 31;
  int wave = threadIdx.x >> 5;              // 0..7
  int et   = blockIdx.x * 8 + wave;         // e-tile 0..15
  int bn   = blockIdx.y;                    // 0..511
  int m    = lane & 15;                     // A row / B,D column within tile
  int half = lane >> 4;

  // Per-thread async-fill mapping: 1024 16-byte pieces per tile, 4 per thread.
  // Global piece (j, k8): W2T[(nt*16+j)*512 + k8*8 ..], 16 bytes.
  // LDS (fragment-swizzled): frag kc, slot (h*16+j), 16 bf16 per slot:
  //   elem = kc*512 + (h*16+j)*16 + w0
  unsigned shb = (unsigned)(uintptr_t)&Bsh[0][0];
  unsigned ldsoff[4], goff0[4];
#pragma unroll
  for (int q = 0; q < 4; ++q) {
    int p = threadIdx.x + 256 * q;
    int j = p >> 6, k8 = p & 63;
    int kc = k8 >> 2, h = (k8 >> 1) & 1, w0 = (k8 & 1) * 8;
    ldsoff[q] = (unsigned)((kc * 512 + (h * 16 + j) * 16 + w0) * 2);
    goff0[q]  = (unsigned)(j * 1024 + k8 * 16);
  }
  // prefill tiles 0 (buf0) and 1 (buf1): 4+4 async issues per wave
#pragma unroll
  for (int q = 0; q < 4; ++q) async_b128(shb + ldsoff[q], goff0[q], W2T);
#pragma unroll
  for (int q = 0; q < 4; ++q) async_b128(shb + 16384u + ldsoff[q], goff0[q] + 16384u, W2T);

  // Build A fragments (overlaps with async fills). Contiguous K layout (matches B):
  // lane(m,half): idx 0..15 -> K = 32*kc + 16*half + idx
  const float* hprow = hp + bn * HID;
  const float* eprow = ep + (et * 16 + m) * HID;
  v16bf af[16];
#pragma unroll
  for (int kc = 0; kc < 16; ++kc) {
#pragma unroll
    for (int i = 0; i < 16; ++i) {
      int k = kc * 32 + (half << 4) + i;
      float v = hprow[k] + eprow[k];
      v = v > 0.f ? v : 0.f;
      af[kc][i] = (__bf16)v;
    }
  }

  float racc[8];
#pragma unroll
  for (int v = 0; v < 8; ++v) racc[v] = 0.f;

#pragma unroll 1
  for (int nt = 0; nt < 32; ++nt) {
    int buf = nt & 1;
    // tile nt done when at most the 4 issues of tile nt+1 remain (in-order)
    if (nt < 31) asm volatile("s_wait_asynccnt 0x4" ::: "memory");
    else         asm volatile("s_wait_asynccnt 0x0" ::: "memory");
    __syncthreads();

    // B fragments from LDS: frag kc = 32 contiguous bytes at kc*1024 + lane*32.
    // Distance-2 register rotation keeps ~4 ds_loads in flight so each WMMA
    // only needs a partial s_wait_dscnt instead of a full drain.
    const __bf16* bbase = &Bsh[buf][lane * 16];
    v16bf bcur  = *(const v16bf*)(bbase);
    v16bf bnext = *(const v16bf*)(bbase + 512);
    v8f c = {};
#pragma unroll
    for (int kc = 0; kc < 16; ++kc) {
      v16bf bpref = (kc + 2 < 16) ? *(const v16bf*)(bbase + (kc + 2) * 512) : bnext;
      c = __builtin_amdgcn_wmma_f32_16x16x32_bf16(false, af[kc], false, bcur,
                                                  (short)0, c, false, false);
      bcur  = bnext;
      bnext = bpref;
    }

    int n = nt * 16 + m;
    float b2v = b2[n];
    float w3v = W3[n];
#pragma unroll
    for (int v = 0; v < 8; ++v) {
      float y = c[v] + b2v;
      y = y > 0.f ? y : 0.f;
      racc[v] += y * w3v;                 // D row = v + 8*half, col = n
    }

    __syncthreads();                      // everyone done reading buf
    if (nt + 2 < 32) {                    // refill buf with tile nt+2
      unsigned bufb  = shb + (unsigned)(buf * 16384);
      unsigned gtile = (unsigned)((nt + 2) * 16384);
#pragma unroll
      for (int q = 0; q < 4; ++q) async_b128(bufb + ldsoff[q], gtile + goff0[q], W2T);
    }
  }

  // reduce over the 16 lanes of each half (columns)
#pragma unroll
  for (int mask = 1; mask <= 8; mask <<= 1) {
#pragma unroll
    for (int v = 0; v < 8; ++v) racc[v] += __shfl_xor(racc[v], mask, 32);
  }

  if (m == 0) {
    float bias = b3[0];
    int ebase = et * 16 + half * 8;       // half 0 -> rows 0..7, half 1 -> 8..15
#pragma unroll
    for (int v = 0; v < 8; ++v) out[bn * NE + ebase + v] = racc[v] + bias;
  }
}

extern "C" void kernel_launch(void* const* d_in, const int* in_sizes, int n_in,
                              void* d_out, int out_size, void* d_ws, size_t ws_size,
                              hipStream_t stream) {
  const float* h_all  = (const float*)d_in[0];
  const float* e_feat = (const float*)d_in[1];
  const float* W1     = (const float*)d_in[2];
  const float* b1     = (const float*)d_in[3];
  const float* W2     = (const float*)d_in[4];
  const float* b2     = (const float*)d_in[5];
  const float* W3     = (const float*)d_in[6];
  const float* b3     = (const float*)d_in[7];
  float* out = (float*)d_out;

  // workspace layout: hp (1MB) | ep (512KB) | W2T bf16 (512KB)
  float*  hp  = (float*)d_ws;
  float*  ep  = hp + BN * HID;
  __bf16* W2T = (__bf16*)(ep + NE * HID);

  prep_hpart<<<BN / 8, 256, 0, stream>>>(h_all, W1, hp);
  prep_epart<<<NE / 8, 256, 0, stream>>>(e_feat, W1, b1, ep);
  prep_w2t<<<(HID * HID) / 256, 256, 0, stream>>>(W2, W2T);
  fused_mlp<<<dim3(2, BN), 256, 0, stream>>>(hp, ep, W2T, b2, W3, b3, out);
}
